// LlamaDLODecoderLayer_17858474017182
// MI455X (gfx1250) — matrix operand order repaired and checked by reading, not compiled, not run on hardware
//
#include <hip/hip_runtime.h>
#include <hip/hip_bf16.h>
#include <stdint.h>

// ---------------------------------------------------------------------------
// Problem constants (match reference)
// ---------------------------------------------------------------------------
#define Bq   4
#define Sq   8192
#define Hq   2048
#define NHq  16
#define DHq  128
#define DFFq 8192
#define Kq   1024
#define EPSq 1e-5f
#define BKq  (Bq * Kq)          // 4096 rows through the layer

typedef __attribute__((ext_vector_type(16))) _Float16 v16h;
typedef __attribute__((ext_vector_type(8)))  float    v8f;
typedef __attribute__((ext_vector_type(4)))  unsigned int u32x4;
typedef __attribute__((ext_vector_type(8)))  int      i32x8;
typedef __attribute__((ext_vector_type(4)))  int      i32x4;

union Frag16 {                   // 16 f16 operand fragment = 8 VGPRs
    v16h     v;
    uint4    q[2];
    _Float16 e[16];
};

#define HAS_TDM __has_builtin(__builtin_amdgcn_tensor_load_to_lds)

// ---------------------------------------------------------------------------
// TDM: DMA one 2D tile (nrows x rowlen f16, row stride = stride_elems) from
// global memory into LDS at byte offset lds_off.  D# per CDNA5 ISA 8.3/8.4.
// ---------------------------------------------------------------------------
__device__ __forceinline__ void tdm_load_tile_f16(unsigned lds_off, const _Float16* gsrc,
                                                  unsigned rowlen, unsigned nrows,
                                                  unsigned stride_elems) {
#if HAS_TDM
    const unsigned long long ga = (unsigned long long)(uintptr_t)gsrc;
    u32x4 g0;
    g0[0] = 1u;                                             // count=1 (valid), user mode
    g0[1] = lds_off;                                        // lds_addr (bytes)
    g0[2] = (unsigned)(ga & 0xffffffffu);                   // global_addr[31:0]
    g0[3] = (unsigned)((ga >> 32) & 0x1ffffffu) | (2u << 30); // addr[56:32] | type=2
    const unsigned td0 = 0x40000000u, td1 = 0x40000000u;    // huge tensor dims (no OOB clip)
    i32x8 g1;
    g1[0] = (int)(1u << 16);                                // data_size=1 -> 2 bytes/elem
    g1[1] = (int)((td0 & 0xffffu) << 16);                   // [63:48] tensor_dim0.lo16
    g1[2] = (int)(((td0 >> 16) & 0xffffu) | ((td1 & 0xffffu) << 16));
    g1[3] = (int)(((td1 >> 16) & 0xffffu) | (rowlen << 16)); // [127:112] tile_dim0
    g1[4] = (int)(nrows & 0xffffu);                         // [143:128] tile_dim1 (dim2=0)
    g1[5] = (int)stride_elems;                              // tensor_dim0_stride lo32
    g1[6] = 0;                                              // stride hi16 | dim1_stride lo16
    g1[7] = 0;
    i32x4 g2 = {0, 0, 0, 0}, g3 = {0, 0, 0, 0};             // unused (2D tensor)
  #if defined(__clang_major__) && (__clang_major__ >= 23)
    i32x8 g4 = {0, 0, 0, 0, 0, 0, 0, 0};
    __builtin_amdgcn_tensor_load_to_lds(g0, g1, g2, g3, g4, 0);
  #else
    __builtin_amdgcn_tensor_load_to_lds(g0, g1, g2, g3, 0);
  #endif
#else
    (void)lds_off; (void)gsrc; (void)rowlen; (void)nrows; (void)stride_elems;
#endif
}

// ---------------------------------------------------------------------------
// Small helpers
// ---------------------------------------------------------------------------
__device__ __forceinline__ float half_reduce_max(float v) {
    #pragma unroll
    for (int o = 8; o >= 1; o >>= 1) v = fmaxf(v, __shfl_xor(v, o, 32));
    return v;
}
__device__ __forceinline__ float half_reduce_sum(float v) {
    #pragma unroll
    for (int o = 8; o >= 1; o >>= 1) v += __shfl_xor(v, o, 32);
    return v;
}

// ---------------------------------------------------------------------------
// 1) full copy of hidden_states -> out (float4 vectorized)
// ---------------------------------------------------------------------------
__global__ void k_copy4(const float4* __restrict__ in, float4* __restrict__ out, size_t n4) {
    size_t i = (size_t)blockIdx.x * blockDim.x + threadIdx.x;
    if (i < n4) out[i] = in[i];
}

// ---------------------------------------------------------------------------
// 2) ordered compaction of topk_mask -> idx[B][K], gathered position ids
// ---------------------------------------------------------------------------
__global__ void k_topk(const unsigned char* __restrict__ mask,
                       const int* __restrict__ pos_ids,
                       int* __restrict__ idx, int* __restrict__ posg) {
    __shared__ int sdata[256];
    __shared__ int sbase;
    const int b = blockIdx.x, t = threadIdx.x;
    if (t == 0) sbase = 0;
    __syncthreads();
    for (int c0 = 0; c0 < Sq; c0 += 256) {
        const int i = c0 + t;
        const int f = mask[(size_t)b * Sq + i] ? 1 : 0;
        sdata[t] = f;
        __syncthreads();
        #pragma unroll
        for (int off = 1; off < 256; off <<= 1) {
            int v = (t >= off) ? sdata[t - off] : 0;
            __syncthreads();
            sdata[t] += v;
            __syncthreads();
        }
        const int rank = sbase + sdata[t] - 1;
        if (f && rank < Kq) {
            idx [(size_t)b * Kq + rank] = i;
            posg[(size_t)b * Kq + rank] = pos_ids[(size_t)b * Sq + i];
        }
        __syncthreads();
        if (t == 255) sbase += sdata[255];
        __syncthreads();
    }
}

// ---------------------------------------------------------------------------
// 3a) generic f32 -> f16 convert
// ---------------------------------------------------------------------------
__global__ void k_cvt16(const float* __restrict__ in, _Float16* __restrict__ out, size_t n) {
    size_t i = (size_t)blockIdx.x * blockDim.x + threadIdx.x;
    if (i < n) out[i] = (_Float16)in[i];
}

// ---------------------------------------------------------------------------
// 3b) fused f32 -> f16 convert + transpose:  in[Kd][N] -> out[N][Kd]
//     (weights become n-major so GEMM B tiles are row-contiguous)
// ---------------------------------------------------------------------------
__global__ void __launch_bounds__(256)
k_cvtT(const float* __restrict__ in, _Float16* __restrict__ out, int Kd, int N) {
    __shared__ _Float16 tile[32][33];
    const int k0 = blockIdx.y * 32, n0 = blockIdx.x * 32;
    const int tx = threadIdx.x & 31, ty = threadIdx.x >> 5;   // 32 x 8
    #pragma unroll
    for (int i = 0; i < 32; i += 8)
        tile[ty + i][tx] = (_Float16)in[(size_t)(k0 + ty + i) * N + n0 + tx];
    __syncthreads();
    #pragma unroll
    for (int i = 0; i < 32; i += 8)
        out[(size_t)(n0 + ty + i) * Kd + k0 + tx] = tile[tx][ty + i];
}

// ---------------------------------------------------------------------------
// 4) gather selected rows + RMSNorm(ln1) ; residual f32, normed f16
// ---------------------------------------------------------------------------
__global__ void __launch_bounds__(256)
k_gather_norm(const float* __restrict__ hs, const int* __restrict__ idx,
              const float* __restrict__ lnw,
              float* __restrict__ resid, _Float16* __restrict__ normed) {
    __shared__ float red[256];
    const int row = blockIdx.x;           // 0..BK-1
    const int b = row / Kq;
    const int src = idx[row];
    const float* x = hs + ((size_t)b * Sq + src) * Hq;
    float v[8]; float ss = 0.f;
    #pragma unroll
    for (int j = 0; j < 8; ++j) {
        const int c = threadIdx.x + j * 256;
        v[j] = x[c];
        ss += v[j] * v[j];
    }
    red[threadIdx.x] = ss; __syncthreads();
    for (int o = 128; o >= 1; o >>= 1) {
        if (threadIdx.x < o) red[threadIdx.x] += red[threadIdx.x + o];
        __syncthreads();
    }
    const float r = rsqrtf(red[0] / (float)Hq + EPSq);
    #pragma unroll
    for (int j = 0; j < 8; ++j) {
        const int c = threadIdx.x + j * 256;
        resid [(size_t)row * Hq + c] = v[j];
        normed[(size_t)row * Hq + c] = (_Float16)(v[j] * r * lnw[c]);
    }
}

// ---------------------------------------------------------------------------
// 5) WMMA GEMM: C[M,N](f32|f16) = A[M,Kd]f16 @ B[Kd,N]f16 with B given
//    PRE-TRANSPOSED as Bt[N][Kd].  Block 256 thr = 8 waves; block tile 64x64;
//    wave tile 16x32.  A/B tiles staged into LDS by the Tensor Data Mover
//    (tensor_load_to_lds + s_wait_tensorcnt) when available.
// ---------------------------------------------------------------------------
template<bool OUT_F16>
__global__ void __launch_bounds__(256)
k_gemm(const _Float16* __restrict__ A, const _Float16* __restrict__ Bt,
       void* __restrict__ C, int M, int N, int Kd) {
    __shared__ __align__(16) _Float16 smem[2 * 64 * 32];
    _Float16* As = smem;                 // [m][k] 64x32  @ LDS byte offset 0
    _Float16* Bs = smem + 64 * 32;       // [n][k] 64x32  @ LDS byte offset 4096
    const int t    = threadIdx.x;
    const int lane = t & 31, w = t >> 5;
    const int wm = w >> 1, wn = w & 1;
    const int m0 = blockIdx.y * 64, n0 = blockIdx.x * 64;
    const int half = lane >> 4, nl = lane & 15;
    const int srow = t >> 2, scol = (t & 3) * 8;  // fallback staging: 64 rows x 4 x 8

    v8f c0 = {}; v8f c1 = {};

    for (int k0 = 0; k0 < Kd; k0 += 32) {
#if HAS_TDM
        if (w == 0) {   // one wave issues both tile DMAs (EXEC ignored by TDM)
            tdm_load_tile_f16(0u,    A  + (size_t)m0 * Kd + k0, 32u, 64u, (unsigned)Kd);
            tdm_load_tile_f16(4096u, Bt + (size_t)n0 * Kd + k0, 32u, 64u, (unsigned)Kd);
  #if __has_builtin(__builtin_amdgcn_s_wait_tensorcnt)
            __builtin_amdgcn_s_wait_tensorcnt(0);
  #endif
        }
        __syncthreads();
#else
        {   // manual staging: A and Bt tile rows are both contiguous uint4s
            uint4 av = *reinterpret_cast<const uint4*>(A  + (size_t)(m0 + srow) * Kd + k0 + scol);
            uint4 bv = *reinterpret_cast<const uint4*>(Bt + (size_t)(n0 + srow) * Kd + k0 + scol);
            *reinterpret_cast<uint4*>(&As[srow * 32 + scol]) = av;
            *reinterpret_cast<uint4*>(&Bs[srow * 32 + scol]) = bv;
        }
        __syncthreads();
#endif
        // A fragment: row = wm*16+nl ; contiguous K run [16h, 16h+16)
        Frag16 a;
        const _Float16* ab = &As[(wm * 16 + nl) * 32];
        a.q[0] = *reinterpret_cast<const uint4*>(ab + 16 * half);
        a.q[1] = *reinterpret_cast<const uint4*>(ab + 16 * half + 8);
        // B fragments: col n = tile*16+nl ; contiguous K run [16h, 16h+16)
        Frag16 b0, b1;
        const _Float16* bb0 = &Bs[(wn * 32 +      nl) * 32 + 16 * half];
        const _Float16* bb1 = &Bs[(wn * 32 + 16 + nl) * 32 + 16 * half];
        b0.q[0] = *reinterpret_cast<const uint4*>(bb0);
        b0.q[1] = *reinterpret_cast<const uint4*>(bb0 + 8);
        b1.q[0] = *reinterpret_cast<const uint4*>(bb1);
        b1.q[1] = *reinterpret_cast<const uint4*>(bb1 + 8);

        c0 = __builtin_amdgcn_wmma_f32_16x16x32_f16(false, a.v, false, b0.v, (short)0, c0, false, false);
        c1 = __builtin_amdgcn_wmma_f32_16x16x32_f16(false, a.v, false, b1.v, (short)0, c1, false, false);
        __syncthreads();
    }
    // C/D layout: lane(nl,half) slot r -> row r+8*half, col nl
    #pragma unroll
    for (int r = 0; r < 8; ++r) {
        const size_t m  = (size_t)(m0 + wm * 16 + r + 8 * half);
        const size_t na = (size_t)(n0 + wn * 32 + nl);
        if (OUT_F16) {
            ((_Float16*)C)[m * N + na]      = (_Float16)c0[r];
            ((_Float16*)C)[m * N + na + 16] = (_Float16)c1[r];
        } else {
            ((float*)C)[m * N + na]      = c0[r];
            ((float*)C)[m * N + na + 16] = c1[r];
        }
    }
}

// ---------------------------------------------------------------------------
// 6) RoPE + repack:
//    q -> (B,NH,K,DH) f16   (A-side of scores: row-contiguous over dh)
//    k -> (B,NH,K,DH) f16   (B-side of scores: lane=token, contiguous dh run)
//    v -> (B,NH,DH,K) f16   (B-side of P.V:    lane=dh,    contiguous tok run)
// ---------------------------------------------------------------------------
__global__ void __launch_bounds__(64)
k_rope(const float* __restrict__ qf, const float* __restrict__ kf,
       const float* __restrict__ vf, const int* __restrict__ posg,
       _Float16* __restrict__ q16, _Float16* __restrict__ k16,
       _Float16* __restrict__ vT16) {
    const int row = blockIdx.x;           // 0..BK-1
    const int b = row / Kq, j = row % Kq;
    const int d = threadIdx.x;            // 0..63
    const float p   = (float)posg[row];
    const float inv = __expf(-((float)(2 * d) / (float)DHq) * 9.2103403720f); // ln(1e4)
    const float fr  = p * inv;
    const float cs = __cosf(fr), sn = __sinf(fr);
    #pragma unroll 4
    for (int h = 0; h < NHq; ++h) {
        const size_t src = (size_t)row * Hq + h * DHq;
        const float q1 = qf[src + d], q2 = qf[src + 64 + d];
        const float k1 = kf[src + d], k2 = kf[src + 64 + d];
        const size_t hb = (size_t)(b * NHq + h);
        const size_t td = (hb * Kq + j) * DHq;        // (token, dh) base
        q16[td + d]      = (_Float16)(q1 * cs - q2 * sn);
        q16[td + 64 + d] = (_Float16)(q2 * cs + q1 * sn);
        k16[td + d]      = (_Float16)(k1 * cs - k2 * sn);
        k16[td + 64 + d] = (_Float16)(k2 * cs + k1 * sn);
        vT16[(hb * DHq + d)      * Kq + j] = (_Float16)vf[src + d];
        vT16[(hb * DHq + 64 + d) * Kq + j] = (_Float16)vf[src + 64 + d];
    }
}

// ---------------------------------------------------------------------------
// 7) Flash attention, causal, one wave per (b, h, 16-row q tile).
//    All fragment loads are contiguous b128s thanks to the layouts above.
// ---------------------------------------------------------------------------
__global__ void __launch_bounds__(32)
k_attn(const _Float16* __restrict__ q16, const _Float16* __restrict__ k16,
       const _Float16* __restrict__ vT16, float* __restrict__ o) {
    const int qt = blockIdx.x, h = blockIdx.y, b = blockIdx.z;
    const int lane = threadIdx.x, half = lane >> 4, nl = lane & 15;
    const size_t hb = (size_t)(b * NHq + h);
    const _Float16* Q  = q16  + (hb * Kq + qt * 16) * DHq;
    const _Float16* Kt = k16  + hb * Kq * DHq;         // (token, dh)
    const _Float16* Vt = vT16 + hb * DHq * Kq;         // (dh, token)

    __shared__ __align__(16) _Float16 Pt[16 * 32];

    const float NEG = -3.0e38f;
    float mrun[8], lrun[8];
    v8f zero = {};
    v8f acc[8];
    #pragma unroll
    for (int r = 0; r < 8; ++r) { mrun[r] = NEG; lrun[r] = 0.f; acc[r] = zero; }

    // Q fragments for all 4 dh-chunks, loaded once
    Frag16 qa[4];
    #pragma unroll
    for (int dc = 0; dc < 4; ++dc) {
        const _Float16* qb = Q + nl * DHq + dc * 32;
        qa[dc].q[0] = *reinterpret_cast<const uint4*>(qb + 16 * half);
        qa[dc].q[1] = *reinterpret_cast<const uint4*>(qb + 16 * half + 8);
    }

    const int nkt = (qt * 16) / 32 + 1;        // causal 32-key tiles
    for (int kt = 0; kt < nkt; ++kt) {
        // ---- scores S = Q . K^T : B fragment = contiguous dh run of K row ----
        v8f s0 = zero, s1 = zero;
        #pragma unroll
        for (int dc = 0; dc < 4; ++dc) {
            Frag16 b0, b1;
            const _Float16* kr0 = Kt + (size_t)(kt * 32 + nl)      * DHq + dc * 32 + 16 * half;
            const _Float16* kr1 = Kt + (size_t)(kt * 32 + 16 + nl) * DHq + dc * 32 + 16 * half;
            b0.q[0] = *reinterpret_cast<const uint4*>(kr0);
            b0.q[1] = *reinterpret_cast<const uint4*>(kr0 + 8);
            b1.q[0] = *reinterpret_cast<const uint4*>(kr1);
            b1.q[1] = *reinterpret_cast<const uint4*>(kr1 + 8);
            s0 = __builtin_amdgcn_wmma_f32_16x16x32_f16(false, qa[dc].v, false, b0.v, (short)0, s0, false, false);
            s1 = __builtin_amdgcn_wmma_f32_16x16x32_f16(false, qa[dc].v, false, b1.v, (short)0, s1, false, false);
        }
        // ---- scale + causal mask + online softmax ----
        const float sc = 0.08838834764831845f;  // 1/sqrt(128)
        #pragma unroll
        for (int r = 0; r < 8; ++r) {
            const int rowg = qt * 16 + r + 8 * half;
            const int c0i = kt * 32 + nl, c1i = c0i + 16;
            float x0 = (c0i <= rowg) ? s0[r] * sc : NEG;
            float x1 = (c1i <= rowg) ? s1[r] * sc : NEG;
            const float mx = half_reduce_max(fmaxf(x0, x1));
            const float mn = fmaxf(mrun[r], mx);
            const float corr = __expf(mrun[r] - mn);
            const float p0 = (c0i <= rowg) ? __expf(x0 - mn) : 0.f;
            const float p1 = (c1i <= rowg) ? __expf(x1 - mn) : 0.f;
            lrun[r] = lrun[r] * corr + half_reduce_sum(p0 + p1);
            mrun[r] = mn;
            #pragma unroll
            for (int nt = 0; nt < 8; ++nt) acc[nt][r] *= corr;
            Pt[(r + 8 * half) * 32 + nl]      = (_Float16)p0;
            Pt[(r + 8 * half) * 32 + 16 + nl] = (_Float16)p1;
        }
        __syncthreads();
        // ---- O += P . V : A frag from LDS, B frag = contiguous token run ----
        Frag16 pa;
        const _Float16* pb = &Pt[nl * 32];
        pa.q[0] = *reinterpret_cast<const uint4*>(pb + 16 * half);
        pa.q[1] = *reinterpret_cast<const uint4*>(pb + 16 * half + 8);
        #pragma unroll
        for (int nt = 0; nt < 8; ++nt) {
            Frag16 bvf;
            const _Float16* vr = Vt + (size_t)(nt * 16 + nl) * Kq + kt * 32 + 16 * half;
            bvf.q[0] = *reinterpret_cast<const uint4*>(vr);
            bvf.q[1] = *reinterpret_cast<const uint4*>(vr + 8);
            acc[nt] = __builtin_amdgcn_wmma_f32_16x16x32_f16(false, pa.v, false, bvf.v, (short)0, acc[nt], false, false);
        }
        __syncthreads();
    }
    // ---- normalize and write (token-major (BK, H) f32) ----
    #pragma unroll
    for (int nt = 0; nt < 8; ++nt) {
        #pragma unroll
        for (int r = 0; r < 8; ++r) {
            const int rowg = qt * 16 + r + 8 * half;
            o[((size_t)(b * Kq + rowg)) * Hq + h * DHq + nt * 16 + nl] = acc[nt][r] / lrun[r];
        }
    }
}

// ---------------------------------------------------------------------------
// 8) residual add + RMSNorm(ln2): resid2 f32, normed2 f16
// ---------------------------------------------------------------------------
__global__ void __launch_bounds__(256)
k_addnorm(const float* __restrict__ resid, const float* __restrict__ proj,
          const float* __restrict__ lnw,
          float* __restrict__ resid2, _Float16* __restrict__ normed2) {
    __shared__ float red[256];
    const size_t row = blockIdx.x;
    float v[8]; float ss = 0.f;
    #pragma unroll
    for (int j = 0; j < 8; ++j) {
        const int c = threadIdx.x + j * 256;
        v[j] = resid[row * Hq + c] + proj[row * Hq + c];
        ss += v[j] * v[j];
    }
    red[threadIdx.x] = ss; __syncthreads();
    for (int o = 128; o >= 1; o >>= 1) {
        if (threadIdx.x < o) red[threadIdx.x] += red[threadIdx.x + o];
        __syncthreads();
    }
    const float r = rsqrtf(red[0] / (float)Hq + EPSq);
    #pragma unroll
    for (int j = 0; j < 8; ++j) {
        const int c = threadIdx.x + j * 256;
        resid2 [row * Hq + c] = v[j];
        normed2[row * Hq + c] = (_Float16)(v[j] * r * lnw[c]);
    }
}

// ---------------------------------------------------------------------------
// 9) m = silu(gate) * up  (f16 in, f16 out)
// ---------------------------------------------------------------------------
__global__ void k_silu_mul(const _Float16* __restrict__ g, const _Float16* __restrict__ u,
                           _Float16* __restrict__ m, size_t n) {
    size_t i = (size_t)blockIdx.x * blockDim.x + threadIdx.x;
    if (i < n) {
        const float gf = (float)g[i];
        m[i] = (_Float16)((gf / (1.f + __expf(-gf))) * (float)u[i]);
    }
}

// ---------------------------------------------------------------------------
// 10) scatter: out[b, idx[j], :] = resid2 + down * scale_b
// ---------------------------------------------------------------------------
__global__ void __launch_bounds__(256)
k_scatter(const float* __restrict__ resid2, const float* __restrict__ down,
          const int* __restrict__ idx, const float* __restrict__ scores,
          float* __restrict__ out) {
    const size_t row = blockIdx.x;
    const int b = (int)(row / Kq);
    const int src = idx[row];
    const float scale = 0.5f * 1.0f + (scores[b] - 0.5f) * 1.0f;  // SCALE_FACTOR/GAP = 1
    #pragma unroll
    for (int j = 0; j < 8; ++j) {
        const int c = threadIdx.x + j * 256;
        out[((size_t)b * Sq + src) * Hq + c] =
            resid2[row * Hq + c] + down[row * Hq + c] * scale;
    }
}

// ---------------------------------------------------------------------------
// Host-side launch
// ---------------------------------------------------------------------------
extern "C" void kernel_launch(void* const* d_in, const int* in_sizes, int n_in,
                              void* d_out, int out_size, void* d_ws, size_t ws_size,
                              hipStream_t stream) {
    (void)in_sizes; (void)n_in; (void)out_size; (void)ws_size;
    const float*         hs    = (const float*)d_in[0];
    const int*           pids  = (const int*)d_in[1];
    const unsigned char* mask  = (const unsigned char*)d_in[2];
    const float*         tks   = (const float*)d_in[3];
    const float* Wq = (const float*)d_in[5],  *Wk = (const float*)d_in[6];
    const float* Wv = (const float*)d_in[7],  *Wo = (const float*)d_in[8];
    const float* Wg = (const float*)d_in[9],  *Wu = (const float*)d_in[10];
    const float* Wd = (const float*)d_in[11];
    const float* ln1 = (const float*)d_in[12], *ln2 = (const float*)d_in[13];
    float* out = (float*)d_out;

    // deterministic bump allocator over d_ws
    char* base = (char*)d_ws;
    size_t off = 0;
    auto alloc = [&](size_t bytes) -> char* {
        char* p = base + off;
        off = (off + bytes + 255) & ~(size_t)255;
        return p;
    };
    int*      idx    = (int*)     alloc((size_t)BKq * 4);
    int*      posg   = (int*)     alloc((size_t)BKq * 4);
    float*    resid1 = (float*)   alloc((size_t)BKq * Hq * 4);
    _Float16* norm1  = (_Float16*)alloc((size_t)BKq * Hq * 2);
    _Float16* WqT    = (_Float16*)alloc((size_t)Hq * Hq * 2);     // transposed f16 weights
    _Float16* WkT    = (_Float16*)alloc((size_t)Hq * Hq * 2);
    _Float16* WvT    = (_Float16*)alloc((size_t)Hq * Hq * 2);
    _Float16* WoT    = (_Float16*)alloc((size_t)Hq * Hq * 2);
    _Float16* WgT    = (_Float16*)alloc((size_t)Hq * DFFq * 2);
    _Float16* WuT    = (_Float16*)alloc((size_t)Hq * DFFq * 2);
    _Float16* WdT    = (_Float16*)alloc((size_t)DFFq * Hq * 2);
    float*    qf     = (float*)   alloc((size_t)BKq * Hq * 4);
    float*    kf     = (float*)   alloc((size_t)BKq * Hq * 4);
    float*    vf     = (float*)   alloc((size_t)BKq * Hq * 4);
    _Float16* q16    = (_Float16*)alloc((size_t)BKq * Hq * 2);
    _Float16* k16    = (_Float16*)alloc((size_t)BKq * Hq * 2);
    _Float16* vT16   = (_Float16*)alloc((size_t)BKq * Hq * 2);
    float*    attnf  = (float*)   alloc((size_t)BKq * Hq * 4);
    _Float16* attn16 = (_Float16*)alloc((size_t)BKq * Hq * 2);
    float*    oWof   = (float*)   alloc((size_t)BKq * Hq * 4);
    float*    resid2 = (float*)   alloc((size_t)BKq * Hq * 4);
    _Float16* norm2  = (_Float16*)alloc((size_t)BKq * Hq * 2);
    _Float16* g16    = (_Float16*)alloc((size_t)BKq * DFFq * 2);
    _Float16* u16    = (_Float16*)alloc((size_t)BKq * DFFq * 2);
    _Float16* m16    = (_Float16*)alloc((size_t)BKq * DFFq * 2);
    float*    downf  = (float*)   alloc((size_t)BKq * Hq * 4);

    // 1) out = hidden_states
    {
        const size_t n4 = (size_t)Bq * Sq * Hq / 4;
        k_copy4<<<dim3((unsigned)((n4 + 255) / 256)), 256, 0, stream>>>(
            (const float4*)hs, (float4*)out, n4);
    }
    // 2) top-k compaction
    k_topk<<<Bq, 256, 0, stream>>>(mask, pids, idx, posg);
    // 3) weight convert + transpose (in: [Kd][N] f32 -> out: [N][Kd] f16)
    auto cvtT = [&](const float* src, _Float16* dst, int Kd, int N) {
        k_cvtT<<<dim3(N / 32, Kd / 32), 256, 0, stream>>>(src, dst, Kd, N);
    };
    cvtT(Wq, WqT, Hq, Hq);   cvtT(Wk, WkT, Hq, Hq);
    cvtT(Wv, WvT, Hq, Hq);   cvtT(Wo, WoT, Hq, Hq);
    cvtT(Wg, WgT, Hq, DFFq); cvtT(Wu, WuT, Hq, DFFq);
    cvtT(Wd, WdT, DFFq, Hq);
    // 4) gather + RMSNorm1
    k_gather_norm<<<BKq, 256, 0, stream>>>(hs, idx, ln1, resid1, norm1);
    // 5) Q/K/V projections  (4096 x 2048 x 2048)
    dim3 gProj(Hq / 64, BKq / 64);
    k_gemm<false><<<gProj, 256, 0, stream>>>(norm1, WqT, qf, BKq, Hq, Hq);
    k_gemm<false><<<gProj, 256, 0, stream>>>(norm1, WkT, kf, BKq, Hq, Hq);
    k_gemm<false><<<gProj, 256, 0, stream>>>(norm1, WvT, vf, BKq, Hq, Hq);
    // 6) RoPE + repack
    k_rope<<<BKq, 64, 0, stream>>>(qf, kf, vf, posg, q16, k16, vT16);
    // 7) flash attention
    k_attn<<<dim3(Kq / 16, NHq, Bq), 32, 0, stream>>>(q16, k16, vT16, attnf);
    // 8) Wo projection + residual + RMSNorm2
    {
        const size_t n = (size_t)BKq * Hq;
        k_cvt16<<<dim3((unsigned)((n + 255) / 256)), 256, 0, stream>>>(attnf, attn16, n);
    }
    k_gemm<false><<<gProj, 256, 0, stream>>>(attn16, WoT, oWof, BKq, Hq, Hq);
    k_addnorm<<<BKq, 256, 0, stream>>>(resid1, oWof, ln2, resid2, norm2);
    // 9) SwiGLU MLP
    dim3 gFF(DFFq / 64, BKq / 64);
    k_gemm<true><<<gFF, 256, 0, stream>>>(norm2, WgT, g16, BKq, DFFq, Hq);
    k_gemm<true><<<gFF, 256, 0, stream>>>(norm2, WuT, u16, BKq, DFFq, Hq);
    {
        const size_t n = (size_t)BKq * DFFq;
        k_silu_mul<<<dim3((unsigned)((n + 255) / 256)), 256, 0, stream>>>(g16, u16, m16, n);
    }
    k_gemm<false><<<gProj, 256, 0, stream>>>(m16, WdT, downf, BKq, Hq, DFFq);
    // 10) scaled residual + scatter into out
    k_scatter<<<BKq, 256, 0, stream>>>(resid2, downf, idx, tks, out);
}
